// EquivariantReconstructionLayer_72344429134050
// MI455X (gfx1250) — matrix operand order repaired and checked by reading, not compiled, not run on hardware
//
#include <hip/hip_runtime.h>
#include <cstdint>

typedef __attribute__((ext_vector_type(16))) _Float16 v16h;
typedef __attribute__((ext_vector_type(8)))  float    v8f;

// -------------------------------------------------------------------------
// Kernel 1: convert W2 (f32 [32][8192]) into f16, pre-swizzled into the
// WMMA B-matrix (32x16 f16) per-lane register layout, one 256-dword block
// per 16-column n-tile:  dst[tile*256 + lane*8 + j]
//   lane: n = lane&15, khalf = (lane>>4)*16 ; dword j holds K = khalf+2j, +1
// -------------------------------------------------------------------------
__global__ void __launch_bounds__(256)
preswizzle_w2_kernel(const float* __restrict__ W2, uint32_t* __restrict__ dst) {
    int g   = blockIdx.x * 256 + threadIdx.x;     // 0 .. 131071
    int t   = g >> 8;                             // n-tile index (0..511)
    int rem = g & 255;
    int l   = rem >> 3;                           // lane 0..31
    int j   = rem & 7;                            // dword 0..7
    int col = (t << 4) + (l & 15);                // column of W2 (0..8191)
    int k   = ((l >> 4) << 4) + (j << 1);         // K index 0..30
    union { _Float16 h[2]; uint32_t u; } p;
    p.h[0] = (_Float16)W2[(size_t)k * 8192 + col];
    p.h[1] = (_Float16)W2[(size_t)(k + 1) * 8192 + col];
    dst[g] = p.u;
}

// -------------------------------------------------------------------------
// Kernel 2: fused  norm -> MLP1 -> MLP2(WMMA) -> bilinear contraction.
// grid.x = 8 r-strips (16 r each), grid.y = batch-tile chunks (4 tiles of
// 16 batches per workgroup). 256 threads = 8 waves.
// -------------------------------------------------------------------------
#define BT 4

__global__ void __launch_bounds__(256)
equiv_recon_kernel(const float* __restrict__ vin,
                   const float* __restrict__ W1,
                   const float* __restrict__ b1,
                   const uint32_t* __restrict__ w2s,   // pre-swizzled f16 W2
                   const float* __restrict__ b2,
                   float* __restrict__ out,
                   int numTiles) {
    __shared__ uint32_t w2_lds[16384];     // 64 KB : W2 strip, 64 n-tiles
    __shared__ float    b2_lds[1024];      // 4 KB  : bias strip
    __shared__ float    s_lds[16 * 64];    // 4 KB  : norms
    __shared__ float    T_lds[16 * 390];   // ~24 KB: 6 sym comps, padded row
    __shared__ uint32_t h_lds[16 * 16];    // 1 KB  : h as f16 [16][32]
    __shared__ float    w_lds[8 * 16 * 65];// ~33 KB: per-wave w tile, padded

    const int tid    = threadIdx.x;
    const int rstrip = blockIdx.x;          // 0..7
    const int lane   = tid & 31;
    const int wave   = tid >> 5;
    const int hi     = lane >> 4;           // 0/1
    const int nlo    = lane & 15;

    // ---- stage W2 strip + b2 strip into LDS (once per WG) ----
    {
        const uint32_t* src = w2s + (size_t)rstrip * 16384;
        #pragma unroll
        for (int i = 0; i < 16; ++i) {
            int o = (i * 256 + tid) * 4;
            *(uint4*)&w2_lds[o] = *(const uint4*)&src[o];
        }
        const float* b2p = b2 + (size_t)rstrip * 1024;
        for (int i = tid; i < 1024; i += 256) b2_lds[i] = b2p[i];
    }
    __syncthreads();

    for (int ti = 0; ti < BT; ++ti) {
        int tile = blockIdx.y * BT + ti;
        if (tile >= numTiles) break;
        int b0 = tile * 16;

        // ---- norms + symmetric outer-product components ----
        #pragma unroll
        for (int q = 0; q < 4; ++q) {
            int idx = q * 256 + tid;            // (b_local, c) flat, 1024
            int bl = idx >> 6, c = idx & 63;
            const float* vp = vin + ((size_t)(b0 + bl) * 64 + c) * 3;
            float x = vp[0], y = vp[1], z = vp[2];
            s_lds[bl * 64 + c] = sqrtf(x * x + y * y + z * z);
            float* Tp = &T_lds[bl * 390 + c * 6];
            Tp[0] = x * x; Tp[1] = y * y; Tp[2] = z * z;
            Tp[3] = x * y; Tp[4] = x * z; Tp[5] = y * z;
        }
        __syncthreads();

        // ---- tiny MLP layer 1: h = relu(s @ W1 + b1), stored as f16 ----
        {
            int k  = tid & 31;
            int bl = tid >> 5;                  // 0..7
            #pragma unroll
            for (int half = 0; half < 2; ++half) {
                int b = bl + half * 8;
                float acc = b1[k];
                const float* srow = &s_lds[b * 64];
                #pragma unroll 8
                for (int c = 0; c < 64; ++c)
                    acc = fmaf(srow[c], W1[c * 32 + k], acc);
                acc = fmaxf(acc, 0.0f);
                ((_Float16*)h_lds)[b * 32 + k] = (_Float16)acc;
            }
        }
        __syncthreads();

        // ---- load A (h tile) into WMMA register layout ----
        union { v16h v; uint4 q[2]; } A;
        {
            const uint32_t* hrow = &h_lds[nlo * 16];     // row m = lane&15
            A.q[0] = *(const uint4*)&hrow[hi * 8];       // K 16*hi .. +7
            A.q[1] = *(const uint4*)&hrow[hi * 8 + 4];   // K 16*hi+8 .. +7
        }

        // ---- each wave owns 2 r values of this strip ----
        float* wbuf = &w_lds[wave * 16 * 65];
        #pragma unroll
        for (int rr = 0; rr < 2; ++rr) {
            int r_local = wave * 2 + rr;
            int r = rstrip * 16 + r_local;

            // stage 2: w[16b][64c] = h @ W2 + b2 via 4 WMMAs
            #pragma unroll
            for (int c0t = 0; c0t < 4; ++c0t) {
                int tileIdx = r_local * 4 + c0t;
                union { v16h v; uint4 q[2]; } Bm;
                const uint32_t* bp = &w2_lds[tileIdx * 256 + lane * 8];
                Bm.q[0] = *(const uint4*)&bp[0];
                Bm.q[1] = *(const uint4*)&bp[4];
                float bias = b2_lds[r_local * 64 + c0t * 16 + nlo];
                v8f acc;
                #pragma unroll
                for (int j = 0; j < 8; ++j) acc[j] = bias;
                acc = __builtin_amdgcn_wmma_f32_16x16x32_f16(
                        false, A.v, false, Bm.v, (short)0, acc, false, false);
                // D layout: vgpr j -> batch j + 8*hi ; n = lane&15
                int base = hi * 8 * 65 + (c0t * 16 + nlo);
                #pragma unroll
                for (int j = 0; j < 8; ++j) wbuf[base + j * 65] = acc[j];
            }
            // make wave's own LDS stores visible to its other lanes
            asm volatile("s_wait_dscnt 0" ::: "memory");

            // stage 3: out[b,r,ij] = sum_c w[b,c] * T[b,c,ij]
            int b = nlo;                        // lane pair (hi=0/1) shares b
            const float* wrow = &wbuf[b * 65 + hi * 32];
            const float* Trow = &T_lds[b * 390 + hi * 32 * 6];
            float m0 = 0, m1 = 0, m2 = 0, m3 = 0, m4 = 0, m5 = 0;
            #pragma unroll 4
            for (int c = 0; c < 32; ++c) {
                float wv = wrow[c];
                const float* Tp = &Trow[c * 6];
                float2 t01 = *(const float2*)&Tp[0];
                float2 t23 = *(const float2*)&Tp[2];
                float2 t45 = *(const float2*)&Tp[4];
                m0 = fmaf(wv, t01.x, m0); m1 = fmaf(wv, t01.y, m1);
                m2 = fmaf(wv, t23.x, m2); m3 = fmaf(wv, t23.y, m3);
                m4 = fmaf(wv, t45.x, m4); m5 = fmaf(wv, t45.y, m5);
            }
            m0 += __shfl_xor(m0, 16, 32);
            m1 += __shfl_xor(m1, 16, 32);
            m2 += __shfl_xor(m2, 16, 32);
            m3 += __shfl_xor(m3, 16, 32);
            m4 += __shfl_xor(m4, 16, 32);
            m5 += __shfl_xor(m5, 16, 32);
            if (hi == 0) {
                float* o = out + ((size_t)(b0 + b) * 128 + r) * 9;
                o[0] = m0; o[1] = m3; o[2] = m4;
                o[3] = m3; o[4] = m1; o[5] = m5;
                o[6] = m4; o[7] = m5; o[8] = m2;
            }
        }
        __syncthreads();   // before next tile overwrites s/T/h
    }
}

// -------------------------------------------------------------------------
extern "C" void kernel_launch(void* const* d_in, const int* in_sizes, int n_in,
                              void* d_out, int out_size, void* d_ws, size_t ws_size,
                              hipStream_t stream) {
    (void)n_in; (void)out_size; (void)ws_size;
    const float* v  = (const float*)d_in[0];   // [B,64,3]
    const float* W1 = (const float*)d_in[1];   // [64,32]
    const float* b1 = (const float*)d_in[2];   // [32]
    const float* W2 = (const float*)d_in[3];   // [32,8192]
    const float* b2 = (const float*)d_in[4];   // [8192]
    float* out = (float*)d_out;                // [B,128,3,3]
    uint32_t* w2s = (uint32_t*)d_ws;           // 512 KB pre-swizzled f16 W2

    int B = in_sizes[0] / 192;                 // elements / (64*3)
    int numTiles = B >> 4;                     // 16-batch tiles (B=100000 -> 6250)

    preswizzle_w2_kernel<<<512, 256, 0, stream>>>(W2, w2s);

    int by = (numTiles + BT - 1) / BT;
    equiv_recon_kernel<<<dim3(8, by), 256, 0, stream>>>(v, W1, b1, w2s, b2, out,
                                                        numTiles);
}
